// LGnet_63333587747392
// MI455X (gfx1250) — compile-verified, matrix operand order checked by reading, not run on hardware
//
#include <hip/hip_runtime.h>
#include <cstdint>
#include <cstddef>

// ---------------------------------------------------------------------------
// LGnet recurrent scan for MI455X (gfx1250, wave32, WMMA).
// B=64,T=256,D=H=512,COMB=1536.
// grid = 4 WGs (16 batch rows each, one WMMA M-tile), 512 threads = 16 waves.
// Each wave owns 32 output columns (2 N-tiles) of every GEMM; h,c live in
// registers across the whole T loop. Weights are bf16, pre-packed into the
// native V_WMMA_F32_16X16X32_BF16 B-fragment layout by a prep kernel and
// streamed from (L2-resident) global memory every step.
// ---------------------------------------------------------------------------

typedef __attribute__((ext_vector_type(16))) __bf16 v16bf;
typedef __attribute__((ext_vector_type(8)))  float  v8f;

#define Bsz   64
#define Tlen  256
#define Ddim  512
#define Hdim  512
#define KT_COMB 48                      // 1536 / 32
#define KT_D    16                      // 512 / 32
#define GATE_TILES (32 * 48)            // ntiles * ktiles per gate
#define GATE_STRIDE ((size_t)GATE_TILES * 512)  // ushorts per packed gate
#define GH_TILES   (32 * 16)

__device__ __forceinline__ unsigned short f2bf(float f) {
  union { float f; unsigned int u; } v; v.f = f;
  unsigned int u = v.u;
  u += 0x7fffu + ((u >> 16) & 1u);       // round-to-nearest-even
  return (unsigned short)(u >> 16);
}

__device__ __forceinline__ float sigmoidf_(float x) {
  return 1.0f / (1.0f + __expf(-x));
}

// ---------------------------------------------------------------------------
// Prep: convert fp32 weights [K, N] row-major into bf16 B-fragments.
// Fragment layout (16-bit B, 32x16 tile, wave32):
//   lane L: column n = L%16; lane group L/16 selects K subsets
//   {0..7,16..23} vs {8..15,24..31}; 16 bf16 per lane, K-pairs per dword.
// One block (512 thr) packs one 32x16 tile: thread = L*16 + e.
// ---------------------------------------------------------------------------
__global__ __launch_bounds__(512)
void lgnet_pack(const float* __restrict__ Wi, const float* __restrict__ Wf,
                const float* __restrict__ Wo, const float* __restrict__ Wc,
                const float* __restrict__ ghW,
                unsigned short* __restrict__ pw, unsigned short* __restrict__ pgh)
{
  const int tile = blockIdx.x;
  const int tid  = threadIdx.x;
  const int L    = tid >> 4;            // 0..31
  const int e    = tid & 15;            // 0..15
  const int base = (L >= 16) ? 8 : 0;
  const int d    = e >> 1;
  const int kk   = base + ((d < 4) ? (2 * d) : (16 + 2 * (d - 4))) + (e & 1);
  const int col  = L & 15;

  if (tile < 4 * GATE_TILES) {
    const int g   = tile / GATE_TILES;
    const int rem = tile % GATE_TILES;
    const int nt  = rem / KT_COMB;
    const int kt  = rem % KT_COMB;
    const float* W = (g == 0) ? Wi : (g == 1) ? Wf : (g == 2) ? Wo : Wc;
    const float v = W[(size_t)(kt * 32 + kk) * Hdim + nt * 16 + col];
    pw[(size_t)tile * 512 + tid] = f2bf(v);
  } else {
    const int t2 = tile - 4 * GATE_TILES;
    const int nt = t2 / KT_D;
    const int kt = t2 % KT_D;
    const float v = ghW[(size_t)(kt * 32 + kk) * Ddim + nt * 16 + col];
    pgh[(size_t)t2 * 512 + tid] = f2bf(v);
  }
}

// ---------------------------------------------------------------------------
// Main recurrent kernel. blockIdx.x in [0,4): batch rows [16b, 16b+16).
// ---------------------------------------------------------------------------
__global__ __launch_bounds__(512, 1)
void lgnet_main(const float* __restrict__ x, const float* __restrict__ Xmean,
                const float* __restrict__ bi_g, const float* __restrict__ bf_g,
                const float* __restrict__ bo_g, const float* __restrict__ bc_g,
                const float* __restrict__ gxw_g, const float* __restrict__ gxb_g,
                const float* __restrict__ ghb_g,
                const unsigned short* __restrict__ pw,
                const unsigned short* __restrict__ pgh,
                float* __restrict__ out)
{
  // A-fragment staging: comb = [xt(16kt) | h(16kt) | mask(16kt)], dl = 16kt.
  __shared__ __align__(16) unsigned short combA[KT_COMB * 512];  // 48 KB
  __shared__ __align__(16) unsigned short dlA[KT_D * 512];       // 16 KB

  const int tid  = threadIdx.x;
  const int lane = tid & 31;
  const int w    = tid >> 5;            // wave 0..15
  const int b0   = blockIdx.x << 4;     // batch tile base

  // ---- elementwise constants: this thread owns column k = tid, all 16 rows
  const int k    = tid;
  const float gxw = gxw_g[k];
  const float gxb = gxb_g[k];
  const int kk   = k & 31;
  const int ktEl = k >> 5;
  // A-fragment scatter offset for (m, kk): index = m*16 + grp*256 + dw*2 + half
  const int elo  = ((kk >> 3) & 1) * 256
                 + (((kk & 16) >> 2) | ((kk & 7) >> 1)) * 2
                 + (kk & 1);

  // ---- per-wave GEMM constants: columns n0/n1 of all N=512 outputs
  const int col  = lane & 15;
  const int mofs = (lane >= 16) ? 8 : 0;          // C/D layout: lanes 16-31 -> M+8
  const int n0   = w * 32 + col;
  const int n1   = n0 + 16;
  const float bi0 = bi_g[n0], bi1 = bi_g[n1];
  const float bf0 = bf_g[n0], bf1 = bf_g[n1];
  const float bo0 = bo_g[n0], bo1 = bo_g[n1];
  const float bc0 = bc_g[n0], bc1 = bc_g[n1];
  const float gh0 = ghb_g[n0], gh1 = ghb_g[n1];

  // h-region scatter offsets in combA (kglobal = 512 + n -> ktile 16 + w)
  int hoff[2];
#pragma unroll
  for (int ntl = 0; ntl < 2; ++ntl) {
    const int kkh = ntl * 16 + col;
    hoff[ntl] = (16 + w) * 512
              + ((kkh >> 3) & 1) * 256
              + ((((kkh & 16) >> 2) | ((kkh & 7) >> 1))) * 2
              + (kkh & 1);
  }

  v8f hA = {}, hB = {}, cA = {}, cB = {};   // recurrent state, 2 N-tiles

  const size_t bstride = (size_t)4 * Tlen * Ddim;   // x batch stride
  const size_t cstride = (size_t)Tlen * Ddim;       // x channel stride

  for (int t = 0; t < Tlen; ++t) {
    // LICM defeat: make an integer offset opaque per step so the t-invariant
    // weight-fragment loads cannot be hoisted (and spilled) out of the time
    // loop. The pointers themselves stay derived from the kernel args, so
    // address-space inference still emits global_load (not flat_load).
    size_t zero = 0;
    asm volatile("" : "+s"(zero));
    const unsigned short* pw_t  = pw  + zero;
    const unsigned short* pgh_t = pgh + zero;

    __syncthreads();   // previous step's gate-GEMM reads of combA/dlA done

    // ---- elementwise pass: xt, mask, dl -> bf16 A-fragments in LDS --------
    {
      const float xm = Xmean[(size_t)t * Ddim + k];
      const float* px = x + (size_t)b0 * bstride + (size_t)t * Ddim + k;
      // prefetch next step's rows (global_prefetch_b8) to hide HBM latency
      if (t + 1 < Tlen) {
        __builtin_prefetch(px + Ddim, 0, 0);
        __builtin_prefetch(px + cstride + Ddim, 0, 0);
        __builtin_prefetch(px + 2 * cstride + Ddim, 0, 0);
        __builtin_prefetch(px + 3 * cstride + Ddim, 0, 0);
      }
#pragma unroll 4
      for (int m = 0; m < 16; ++m) {
        const float* p  = px + (size_t)m * bstride;
        const float Xv  = p[0];
        const float Xl  = p[cstride];
        const float Mv  = p[2 * cstride];
        const float Dv  = p[3 * cstride];
        const float dx  = __expf(-fmaxf(Dv * gxw + gxb, 0.0f));
        const float xt  = Mv * Xv + (1.0f - Mv) * (dx * Xl + (1.0f - dx) * xm);
        const int o = m * 16 + elo;
        combA[ktEl * 512 + o]        = f2bf(xt);   // comb[0:512)   = xt
        combA[(32 + ktEl) * 512 + o] = f2bf(Mv);   // comb[1024:)   = mask
        dlA[ktEl * 512 + o]          = f2bf(Dv);   // dl fragments
      }
    }
    __syncthreads();

    // ---- delta_h = exp(-relu(dl @ ghW + ghb)); h *= delta_h; h -> comb ----
    {
      v8f a0 = {}, a1 = {};
#pragma clang loop unroll_count(4)
      for (int kt = 0; kt < KT_D; ++kt) {
        const v16bf A  = *reinterpret_cast<const v16bf*>(&dlA[kt * 512 + lane * 16]);
        const v16bf B0 = *reinterpret_cast<const v16bf*>(
            &pgh_t[((size_t)(w * 2 + 0) * KT_D + kt) * 512 + lane * 16]);
        const v16bf B1 = *reinterpret_cast<const v16bf*>(
            &pgh_t[((size_t)(w * 2 + 1) * KT_D + kt) * 512 + lane * 16]);
        a0 = __builtin_amdgcn_wmma_f32_16x16x32_bf16(false, A, false, B0, (short)0, a0, false, false);
        a1 = __builtin_amdgcn_wmma_f32_16x16x32_bf16(false, A, false, B1, (short)0, a1, false, false);
      }
#pragma unroll
      for (int v = 0; v < 8; ++v) {
        hA[v] *= __expf(-fmaxf(a0[v] + gh0, 0.0f));
        hB[v] *= __expf(-fmaxf(a1[v] + gh1, 0.0f));
        combA[hoff[0] + (v + mofs) * 16] = f2bf(hA[v]);   // comb[512:1024) = h
        combA[hoff[1] + (v + mofs) * 16] = f2bf(hB[v]);
      }
    }
    __syncthreads();

    // ---- 4 gate GEMMs over K=1536, then LSTM cell update ------------------
#pragma unroll
    for (int ntl = 0; ntl < 2; ++ntl) {
      const size_t tb = ((size_t)(w * 2 + ntl) * KT_COMB) * 512 + (size_t)lane * 16;
      v8f gi = {}, gf = {}, go = {}, gc = {};
#pragma clang loop unroll_count(2)
      for (int kt = 0; kt < KT_COMB; ++kt) {
        const v16bf A = *reinterpret_cast<const v16bf*>(&combA[kt * 512 + lane * 16]);
        const unsigned short* bptr = pw_t + tb + (size_t)kt * 512;
        const v16bf Bi = *reinterpret_cast<const v16bf*>(bptr);
        const v16bf Bf = *reinterpret_cast<const v16bf*>(bptr + GATE_STRIDE);
        const v16bf Bo = *reinterpret_cast<const v16bf*>(bptr + 2 * GATE_STRIDE);
        const v16bf Bc = *reinterpret_cast<const v16bf*>(bptr + 3 * GATE_STRIDE);
        gi = __builtin_amdgcn_wmma_f32_16x16x32_bf16(false, A, false, Bi, (short)0, gi, false, false);
        gf = __builtin_amdgcn_wmma_f32_16x16x32_bf16(false, A, false, Bf, (short)0, gf, false, false);
        go = __builtin_amdgcn_wmma_f32_16x16x32_bf16(false, A, false, Bo, (short)0, go, false, false);
        gc = __builtin_amdgcn_wmma_f32_16x16x32_bf16(false, A, false, Bc, (short)0, gc, false, false);
      }
      v8f& cc = ntl ? cB : cA;
      v8f& hh = ntl ? hB : hA;
      const float bii = ntl ? bi1 : bi0;
      const float bff = ntl ? bf1 : bf0;
      const float boo = ntl ? bo1 : bo0;
      const float bcc = ntl ? bc1 : bc0;
      const int n = n0 + ntl * 16;
#pragma unroll
      for (int v = 0; v < 8; ++v) {
        const float ig = sigmoidf_(gi[v] + bii);
        const float fg = sigmoidf_(gf[v] + bff);
        const float og = sigmoidf_(go[v] + boo);
        const float ct = tanhf(gc[v] + bcc);
        const float cn = fg * cc[v] + ig * ct;
        cc[v] = cn;
        const float hn = og * tanhf(cn);
        hh[v] = hn;
        out[((size_t)(b0 + v + mofs) * Tlen + t) * Hdim + n] = hn;
      }
    }
  }
}

// ---------------------------------------------------------------------------
// Inputs (setup_inputs order): 0:x 1:X_mean 2:Wi 3:bi 4:Wf 5:bf 6:Wo 7:bo
//                              8:Wc 9:bc 10:gx_w 11:gx_b 12:gh_W 13:gh_b
// Workspace: 4 packed gates (6.29 MB) + packed ghW (0.52 MB) bf16.
// ---------------------------------------------------------------------------
extern "C" void kernel_launch(void* const* d_in, const int* in_sizes, int n_in,
                              void* d_out, int out_size, void* d_ws, size_t ws_size,
                              hipStream_t stream) {
  (void)in_sizes; (void)n_in; (void)out_size; (void)ws_size;
  const float* x     = (const float*)d_in[0];
  const float* Xmean = (const float*)d_in[1];
  const float* Wi    = (const float*)d_in[2];
  const float* bi    = (const float*)d_in[3];
  const float* Wf    = (const float*)d_in[4];
  const float* bf    = (const float*)d_in[5];
  const float* Wo    = (const float*)d_in[6];
  const float* bo    = (const float*)d_in[7];
  const float* Wc    = (const float*)d_in[8];
  const float* bc    = (const float*)d_in[9];
  const float* gxw   = (const float*)d_in[10];
  const float* gxb   = (const float*)d_in[11];
  const float* ghW   = (const float*)d_in[12];
  const float* ghb   = (const float*)d_in[13];

  unsigned short* pw  = (unsigned short*)d_ws;
  unsigned short* pgh = pw + 4 * GATE_STRIDE;

  const int packTiles = 4 * GATE_TILES + GH_TILES;   // 6656
  lgnet_pack<<<packTiles, 512, 0, stream>>>(Wi, Wf, Wo, Wc, ghW, pw, pgh);
  lgnet_main<<<4, 512, 0, stream>>>(x, Xmean, bi, bf, bo, bc, gxw, gxb, ghb,
                                    pw, pgh, (float*)d_out);
}
